// MultiHeadSelfAttention_35330400977510
// MI455X (gfx1250) — matrix operand order repaired and checked by reading, not compiled, not run on hardware
//
#include <hip/hip_runtime.h>
#include <hip/hip_bf16.h>

#define BSZ 4
#define SEQ 2048
#define DM  1024
#define NH  16
#define DH  64
#define E3  (3*DM)
#define MROWS (BSZ*SEQ)   // 8192

typedef __attribute__((ext_vector_type(16))) __bf16 v16bf;
typedef __attribute__((ext_vector_type(8)))  __bf16 v8bf;
typedef __attribute__((ext_vector_type(8)))  float  v8f;

// ---- WMMA wrapper: D = A(16x32 bf16) * B(32x16 bf16) + C(16x16 f32) ----
__device__ inline v8f wmma_bf16(v16bf a, v16bf b, v8f c) {
  return __builtin_amdgcn_wmma_f32_16x16x32_bf16(
      /*neg_a=*/false, a, /*neg_b=*/false, b,
      /*c_mod=*/(short)0, c, /*reuse_a=*/false, /*reuse_b=*/false);
}

// A fragment (16x32, 16-bit): lane holds row m=lane&15.
// element e: K = 16*(lane>>4) + e  -> one contiguous 16-elem run.
__device__ inline v16bf load_a_frag(const __bf16* base, int row, int ld, int k0, int lane) {
  const __bf16* p = base + (size_t)row * ld + k0 + 16 * (lane >> 4);
  v8bf lo = *(const v8bf*)p;
  v8bf hi = *(const v8bf*)(p + 8);
  v16bf a;
#pragma unroll
  for (int i = 0; i < 8; ++i) { a[i] = lo[i]; a[i + 8] = hi[i]; }
  return a;
}

// B fragment (32x16, 16-bit): lane holds column n=lane&15.
// element e: K = 16*(lane>>4) + e  -> one contiguous 16-elem run.
__device__ inline v16bf load_b_frag(const __bf16* base, int col, int ld, int k0, int lane) {
  const __bf16* p = base + (size_t)col * ld + k0 + 16 * (lane >> 4);
  v8bf lo = *(const v8bf*)p;
  v8bf hi = *(const v8bf*)(p + 8);
  v16bf b;
#pragma unroll
  for (int i = 0; i < 8; ++i) { b[i] = lo[i]; b[i + 8] = hi[i]; }
  return b;
}

// ---------------- fp32 -> bf16 convert ----------------
__global__ __launch_bounds__(256) void cvt_bf16_kernel(const float* __restrict__ in,
                                                       __bf16* __restrict__ out, int n) {
  int i = blockIdx.x * 256 + threadIdx.x;
  if (i < n) out[i] = (__bf16)in[i];
}

// ---------------- QKV GEMM (16x64 per wave, SW-pipelined) + RoPE + head scatter ----
// qkv[m, e] = sum_d xb[m,d] * wqkv[e,d]; q/k get RoPE, q gets 1/sqrt(Dh).
// q,k -> [B,H,S,Dh] bf16 ; v -> transposed [B,H,Dh,S] bf16.
__global__ __launch_bounds__(256) void qkv_rope_kernel(const __bf16* __restrict__ xb,
                                                       const __bf16* __restrict__ wb,
                                                       const int* __restrict__ tok,
                                                       __bf16* __restrict__ qg,
                                                       __bf16* __restrict__ kg,
                                                       __bf16* __restrict__ vTg) {
  const int lane = threadIdx.x & 31;
  const int wave = threadIdx.x >> 5;
  const int n0 = blockIdx.x * 64;                 // 64-wide column strip in [0, 3072)
  const int m0 = (blockIdx.y * 8 + wave) * 16;    // row tile in [0, 8192)

  v8f acc[4];
#pragma unroll
  for (int t = 0; t < 4; ++t) acc[t] = (v8f){0.f,0.f,0.f,0.f,0.f,0.f,0.f,0.f};

  const int arow = m0 + (lane & 15);
  const int bcol = n0 + (lane & 15);
  const __bf16* aprefetch = xb + (size_t)arow * DM;

  // --- software pipeline: double-buffer fragments across k-steps ---
  v16bf a  = load_a_frag(xb, arow, DM, 0, lane);
  v16bf b0 = load_b_frag(wb, bcol +  0, DM, 0, lane);
  v16bf b1 = load_b_frag(wb, bcol + 16, DM, 0, lane);
  v16bf b2 = load_b_frag(wb, bcol + 32, DM, 0, lane);
  v16bf b3 = load_b_frag(wb, bcol + 48, DM, 0, lane);

  for (int k0 = 32; k0 < DM; k0 += 32) {
    __builtin_prefetch(aprefetch + k0 + 256, 0, 3);   // global_prefetch, near scope
    v16bf an  = load_a_frag(xb, arow, DM, k0, lane);
    v16bf bn0 = load_b_frag(wb, bcol +  0, DM, k0, lane);
    v16bf bn1 = load_b_frag(wb, bcol + 16, DM, k0, lane);
    v16bf bn2 = load_b_frag(wb, bcol + 32, DM, k0, lane);
    v16bf bn3 = load_b_frag(wb, bcol + 48, DM, k0, lane);
    acc[0] = wmma_bf16(a, b0, acc[0]);                // overlaps next-step loads
    acc[1] = wmma_bf16(a, b1, acc[1]);
    acc[2] = wmma_bf16(a, b2, acc[2]);
    acc[3] = wmma_bf16(a, b3, acc[3]);
    a = an; b0 = bn0; b1 = bn1; b2 = bn2; b3 = bn3;
  }
  acc[0] = wmma_bf16(a, b0, acc[0]);
  acc[1] = wmma_bf16(a, b1, acc[1]);
  acc[2] = wmma_bf16(a, b2, acc[2]);
  acc[3] = wmma_bf16(a, b3, acc[3]);

  const int rbase = m0 + 8 * (lane >> 4);
#pragma unroll
  for (int t = 0; t < 4; ++t) {
    const int e      = bcol + 16 * t;   // all lanes share part / head within a 16-tile
    const int part   = e >> 10;         // 0=q, 1=k, 2=v
    const int within = e & 1023;
    const int h      = within >> 6;
    const int dh     = within & 63;

    if (part < 2) {
      const float inv     = __powf(10000.0f, -(float)(dh & ~1) / (float)DH);
      const bool  is_even = (dh & 1) == 0;
      const float qscale  = (part == 0) ? 0.125f : 1.0f;
      __bf16* dst = (part == 0) ? qg : kg;
#pragma unroll
      for (int r = 0; r < 8; ++r) {
        const int row = rbase + r;
        const int s   = row & (SEQ - 1);
        const int bi  = row >> 11;                  // row / SEQ
        float val   = acc[t][r];
        float other = __shfl_xor(val, 1, 32);       // partner (even<->odd dh)
        float ang = (float)tok[s] * inv;
        float sn, cs;
        __sincosf(ang, &sn, &cs);
        float out = is_even ? (val * cs - other * sn) : (other * sn + val * cs);
        dst[(((size_t)(bi * NH + h)) * SEQ + s) * DH + dh] = (__bf16)(out * qscale);
      }
    } else {
#pragma unroll
      for (int r = 0; r < 8; ++r) {
        const int row = rbase + r;
        const int s   = row & (SEQ - 1);
        const int bi  = row >> 11;
        vTg[(((size_t)(bi * NH + h)) * DH + dh) * SEQ + s] = (__bf16)acc[t][r];
      }
    }
  }
}

// ---------------- flash attention per (b,h, 16-row q tile) ----------------
// Computes S^T tiles (keys x queries) so each lane owns one query row.
__global__ __launch_bounds__(256) void attn_kernel(const __bf16* __restrict__ qg,
                                                   const __bf16* __restrict__ kg,
                                                   const __bf16* __restrict__ vTg,
                                                   __bf16* __restrict__ attn) {
  const int lane = threadIdx.x & 31;
  const int wave = threadIdx.x >> 5;
  const int bh = blockIdx.y;
  const int bi = bh >> 4;
  const int h  = bh & 15;
  const int q0 = (blockIdx.x * 8 + wave) * 16;

  const __bf16* qb = qg  + (size_t)bh * SEQ * DH;
  const __bf16* kb = kg  + (size_t)bh * SEQ * DH;
  const __bf16* vb = vTg + (size_t)bh * DH * SEQ;

  const int srow = q0 + (lane & 15);              // this lane's query row

  // q B-fragments are loop-invariant (contraction over d = 0..63)
  v16bf bq0 = load_b_frag(qb, srow, DH, 0, lane);
  v16bf bq1 = load_b_frag(qb, srow, DH, 32, lane);

  v8f o0 = {0,0,0,0,0,0,0,0}, o1 = o0, o2 = o0, o3 = o0;  // O^T, 64x16
  float mrun = -1e30f, lrun = 0.f;

  for (int j0 = 0; j0 <= q0 + 15; j0 += 32) {     // 32 keys per step
    // ---- S^T = K_tile x Q^T ---- (hoist 4 K fragments, then 4 WMMAs)
    const int kr0 = j0 + (lane & 15);
    const int kr1 = kr0 + 16;
    v16bf ak00 = load_a_frag(kb, kr0, DH, 0,  lane);
    v16bf ak01 = load_a_frag(kb, kr0, DH, 32, lane);
    v16bf ak10 = load_a_frag(kb, kr1, DH, 0,  lane);
    v16bf ak11 = load_a_frag(kb, kr1, DH, 32, lane);
    v8f s0 = {0,0,0,0,0,0,0,0}, s1 = s0;
    s0 = wmma_bf16(ak00, bq0, s0);
    s0 = wmma_bf16(ak01, bq1, s0);
    s1 = wmma_bf16(ak10, bq0, s1);
    s1 = wmma_bf16(ak11, bq1, s1);

    // ---- causal mask + online softmax (per-lane query row) ----
    const int jb0 = j0 + 8 * (lane >> 4);
    const int jb1 = jb0 + 16;
    float p0[8], p1[8];
    float cm = -1e30f;
#pragma unroll
    for (int r = 0; r < 8; ++r) {
      float a0 = ((jb0 + r) <= srow) ? s0[r] : -1e30f;
      float a1 = ((jb1 + r) <= srow) ? s1[r] : -1e30f;
      p0[r] = a0; p1[r] = a1;
      cm = fmaxf(cm, fmaxf(a0, a1));
    }
    cm = fmaxf(cm, __shfl_xor(cm, 16, 32));       // other 16 keys of the tile
    const float mnew  = fmaxf(mrun, cm);
    const float alpha = __expf(mrun - mnew);
    float lsum = 0.f;
#pragma unroll
    for (int r = 0; r < 8; ++r) {
      p0[r] = __expf(p0[r] - mnew);
      p1[r] = __expf(p1[r] - mnew);
      lsum += p0[r] + p1[r];
    }
    lsum += __shfl_xor(lsum, 16, 32);
    lrun = lrun * alpha + lsum;
    mrun = mnew;

    // ---- V fragment loads issued ahead of the softmax repack ----
    const int vr = lane & 15;
    v16bf av0 = load_a_frag(vb,  0 + vr, SEQ, j0, lane);
    v16bf av1 = load_a_frag(vb, 16 + vr, SEQ, j0, lane);
    v16bf av2 = load_a_frag(vb, 32 + vr, SEQ, j0, lane);
    v16bf av3 = load_a_frag(vb, 48 + vr, SEQ, j0, lane);

#pragma unroll
    for (int r = 0; r < 8; ++r) { o0[r] *= alpha; o1[r] *= alpha; o2[r] *= alpha; o3[r] *= alpha; }

    // ---- repack P^T into B fragment (column n = query row) ----
    float q0o[8], q1o[8];
#pragma unroll
    for (int r = 0; r < 8; ++r) {
      q0o[r] = __shfl_xor(p0[r], 16, 32);
      q1o[r] = __shfl_xor(p1[r], 16, 32);
    }
    v16bf bp;
    if (lane < 16) {
#pragma unroll
      for (int i = 0; i < 8; ++i) { bp[i] = (__bf16)p0[i];  bp[i + 8] = (__bf16)q0o[i]; }
    } else {
#pragma unroll
      for (int i = 0; i < 8; ++i) { bp[i] = (__bf16)q1o[i]; bp[i + 8] = (__bf16)p1[i]; }
    }

    // ---- O^T += V^T_tile x P^T ----  (A rows = d, contiguous in vT layout)
    o0 = wmma_bf16(av0, bp, o0);
    o1 = wmma_bf16(av1, bp, o1);
    o2 = wmma_bf16(av2, bp, o2);
    o3 = wmma_bf16(av3, bp, o3);
  }

  // ---- normalize + store to attn [B,S,D] bf16 (8 contiguous elems/lane) ----
  const float inv = 1.0f / lrun;
  __bf16* outp = attn + ((size_t)(bi * SEQ + srow)) * DM + h * DH + 8 * (lane >> 4);
  v8bf st;
#pragma unroll
  for (int r = 0; r < 8; ++r) st[r] = (__bf16)(o0[r] * inv);
  *(v8bf*)(outp + 0) = st;
#pragma unroll
  for (int r = 0; r < 8; ++r) st[r] = (__bf16)(o1[r] * inv);
  *(v8bf*)(outp + 16) = st;
#pragma unroll
  for (int r = 0; r < 8; ++r) st[r] = (__bf16)(o2[r] * inv);
  *(v8bf*)(outp + 32) = st;
#pragma unroll
  for (int r = 0; r < 8; ++r) st[r] = (__bf16)(o3[r] * inv);
  *(v8bf*)(outp + 48) = st;
}

// ---------------- output projection (16x64 per wave, SW-pipelined) ----------------
__global__ __launch_bounds__(256) void outproj_kernel(const __bf16* __restrict__ attn,
                                                      const __bf16* __restrict__ wb,
                                                      float* __restrict__ out) {
  const int lane = threadIdx.x & 31;
  const int wave = threadIdx.x >> 5;
  const int n0 = blockIdx.x * 64;
  const int m0 = (blockIdx.y * 8 + wave) * 16;

  v8f acc[4];
#pragma unroll
  for (int t = 0; t < 4; ++t) acc[t] = (v8f){0.f,0.f,0.f,0.f,0.f,0.f,0.f,0.f};

  const int arow = m0 + (lane & 15);
  const int bcol = n0 + (lane & 15);
  const __bf16* aprefetch = attn + (size_t)arow * DM;

  v16bf a  = load_a_frag(attn, arow, DM, 0, lane);
  v16bf b0 = load_b_frag(wb, bcol +  0, DM, 0, lane);
  v16bf b1 = load_b_frag(wb, bcol + 16, DM, 0, lane);
  v16bf b2 = load_b_frag(wb, bcol + 32, DM, 0, lane);
  v16bf b3 = load_b_frag(wb, bcol + 48, DM, 0, lane);

  for (int k0 = 32; k0 < DM; k0 += 32) {
    __builtin_prefetch(aprefetch + k0 + 256, 0, 3);
    v16bf an  = load_a_frag(attn, arow, DM, k0, lane);
    v16bf bn0 = load_b_frag(wb, bcol +  0, DM, k0, lane);
    v16bf bn1 = load_b_frag(wb, bcol + 16, DM, k0, lane);
    v16bf bn2 = load_b_frag(wb, bcol + 32, DM, k0, lane);
    v16bf bn3 = load_b_frag(wb, bcol + 48, DM, k0, lane);
    acc[0] = wmma_bf16(a, b0, acc[0]);
    acc[1] = wmma_bf16(a, b1, acc[1]);
    acc[2] = wmma_bf16(a, b2, acc[2]);
    acc[3] = wmma_bf16(a, b3, acc[3]);
    a = an; b0 = bn0; b1 = bn1; b2 = bn2; b3 = bn3;
  }
  acc[0] = wmma_bf16(a, b0, acc[0]);
  acc[1] = wmma_bf16(a, b1, acc[1]);
  acc[2] = wmma_bf16(a, b2, acc[2]);
  acc[3] = wmma_bf16(a, b3, acc[3]);

  const int rbase = m0 + 8 * (lane >> 4);
#pragma unroll
  for (int t = 0; t < 4; ++t)
#pragma unroll
    for (int r = 0; r < 8; ++r)
      out[(size_t)(rbase + r) * DM + bcol + 16 * t] = acc[t][r];
}

extern "C" void kernel_launch(void* const* d_in, const int* in_sizes, int n_in,
                              void* d_out, int out_size, void* d_ws, size_t ws_size,
                              hipStream_t stream) {
  const float* x    = (const float*)d_in[0];
  const int*   tok  = (const int*)  d_in[1];
  const float* wqkv = (const float*)d_in[2];
  const float* wout = (const float*)d_in[3];
  float* out = (float*)d_out;

  // workspace carve-out (bf16 buffers), ~88 MiB total
  char* w = (char*)d_ws;
  __bf16* xb    = (__bf16*)w; w += (size_t)MROWS * DM * 2;
  __bf16* wqkvb = (__bf16*)w; w += (size_t)E3 * DM * 2;
  __bf16* woutb = (__bf16*)w; w += (size_t)DM * DM * 2;
  __bf16* qg    = (__bf16*)w; w += (size_t)MROWS * DM * 2;
  __bf16* kg    = (__bf16*)w; w += (size_t)MROWS * DM * 2;
  __bf16* vTg   = (__bf16*)w; w += (size_t)MROWS * DM * 2;
  __bf16* attn  = (__bf16*)w; w += (size_t)MROWS * DM * 2;

  const int nx = MROWS * DM, nq = E3 * DM, no = DM * DM;
  cvt_bf16_kernel<<<dim3((nx + 255) / 256), dim3(256), 0, stream>>>(x, xb, nx);
  cvt_bf16_kernel<<<dim3((nq + 255) / 256), dim3(256), 0, stream>>>(wqkv, wqkvb, nq);
  cvt_bf16_kernel<<<dim3((no + 255) / 256), dim3(256), 0, stream>>>(wout, woutb, no);

  // QKV projection + RoPE: 48 x 64-wide column strips, 64 row blocks (8 waves/block)
  qkv_rope_kernel<<<dim3(E3 / 64, MROWS / 128), dim3(256), 0, stream>>>(
      xb, wqkvb, tok, qg, kg, vTg);

  // attention: 16 q-tile blocks x 64 (b,h) pairs
  attn_kernel<<<dim3(SEQ / 128, BSZ * NH), dim3(256), 0, stream>>>(qg, kg, vTg, attn);

  // output projection
  outproj_kernel<<<dim3(DM / 64, MROWS / 128), dim3(256), 0, stream>>>(attn, woutb, out);
}
